// NervPlusBlock_21191368638995
// MI455X (gfx1250) — compile-verified, matrix operand order checked by reading, not run on hardware
//
#include <hip/hip_runtime.h>
#include <math.h>

// ---------------- CDNA5 (gfx1250) types / WMMA helpers ----------------
typedef __bf16 bf16;
typedef __attribute__((ext_vector_type(16))) __bf16 v16bf;
typedef __attribute__((ext_vector_type(8)))  float  v8f;
typedef __attribute__((ext_vector_type(4)))  unsigned int v4u;
typedef __attribute__((ext_vector_type(8)))  int v8i;
typedef __attribute__((ext_vector_type(4)))  int v4i;

#define LANE ((int)(threadIdx.x & 31))
#define WAVE ((int)(threadIdx.x >> 5))

#if __has_builtin(__builtin_amdgcn_tensor_load_to_lds) && __has_builtin(__builtin_amdgcn_s_wait_tensorcnt)
#define HAVE_TDM 1
#endif

__device__ __forceinline__ bf16 f2bf(float f){ return (bf16)f; }

__device__ __forceinline__ v8f wmma_bf(v16bf a, v16bf b, v8f c){
  // D(16x16,f32) = A(16x32,bf16) * B(32x16,bf16) + C
  return __builtin_amdgcn_wmma_f32_16x16x32_bf16(false, a, false, b, (short)0, c, false, false);
}

// A fragment from row-major A (lda in elems): two contiguous 16B loads per lane.
__device__ __forceinline__ v16bf load_frag_a(const bf16* A, int lda, int k0, int lane){
  const bf16* p = A + (size_t)(lane & 15) * lda + (k0 + ((lane >> 4) << 3));
  v16bf a;
#pragma unroll
  for (int i = 0; i < 8; i++){ a[i] = p[i]; a[i+8] = p[16 + i]; }
  return a;
}

// B fragment from fragment-packed buffer: [tile][lane][16] -> one 32B contiguous run per lane.
__device__ __forceinline__ v16bf load_frag_b_pk(const bf16* Bpk, int ktiles, int nt, int kt, int lane){
  const bf16* p = Bpk + (((size_t)nt * ktiles + kt) * 32 + lane) * 16;
  v16bf b;
#pragma unroll
  for (int i = 0; i < 16; i++) b[i] = p[i];
  return b;
}

__device__ __forceinline__ float gelu_f(float v){
  return 0.5f * v * (1.f + tanhf(0.7978845608f * (v + 0.044715f * v * v * v)));
}
__device__ __forceinline__ float silu_f(float v){ return v / (1.f + __expf(-v)); }

// ---------------- problem constants ----------------
// B=8 C=256 H=64 W=128 P=4 E=128 HP=16 WP=32 N=512 DM=128 DI=256 DS=16 DC=4 DR=8 NG=128 S=2

// ---------------- weight packing into WMMA B-fragment layout ----------------
// Logical B[k][n] = src[n*Ksrc + k]; zero-pad k>=Ksrc or n>=Nsrc.
__global__ void k_pack_fragB(bf16* dst, const float* src, int K, int Ksrc, int N, int Nsrc){
  int i = blockIdx.x * 256 + threadIdx.x;
  if (i >= K * N) return;
  int e = i & 15, lane = (i >> 4) & 31;
  int ktiles = K >> 5;
  int tile = i >> 9;
  int kt = tile % ktiles, nt = tile / ktiles;
  int n = nt * 16 + (lane & 15);
  int kb = kt * 32 + ((lane >> 4) << 3);
  int k = kb + ((e & 8) ? (16 + (e & 7)) : e);
  float v = (k < Ksrc && n < Nsrc) ? src[(size_t)n * Ksrc + k] : 0.f;
  dst[i] = f2bf(v);
}

// conv weights, K-order permuted to k' = pos*128 + c (pos = ky*3+kx), K=1152 (36 kt), N=512
__global__ void k_pack_conv_fragB(bf16* dst, const float* src){
  int i = blockIdx.x * 256 + threadIdx.x;
  if (i >= 1152 * 512) return;
  int e = i & 15, lane = (i >> 4) & 31;
  int tile = i >> 9;
  int kt = tile % 36, nt = tile / 36;
  int oc = nt * 16 + (lane & 15);
  int kb = kt * 32 + ((lane >> 4) << 3);
  int kp = kb + ((e & 8) ? (16 + (e & 7)) : e);
  int c = kp & 127, pos = kp >> 7;
  dst[i] = f2bf(src[(size_t)oc * 1152 + c * 9 + pos]);
}

__global__ void k_convert_bf16(bf16* dst, const float* src, int n){
  int i = blockIdx.x * 256 + threadIdx.x;
  if (i < n) dst[i] = f2bf(src[i]);
}

// ---------------- stage 1: patch embed (implicit GEMM, per-batch M=512,N=128,K=4096) ----------------
__global__ void k_patch_embed(bf16* t_bf, const float* x, const bf16* Bw, const float* patch_b){
  int tile = blockIdx.x * 8 + WAVE;            // 8 * 32 * 8 = 2048 tiles
  int b = tile >> 8, mt = (tile >> 3) & 31, nt = tile & 7;
  int lane = LANE;
  int mm = mt * 16 + (lane & 15);
  int hp = mm >> 5, wp = mm & 31;              // n = hp*32+wp
  v8f acc = {};
  for (int k0 = 0; k0 < 4096; k0 += 32){
    int kb = k0 + ((lane >> 4) << 3);
    v16bf a;
#pragma unroll
    for (int i = 0; i < 8; i++){
      int k1 = kb + i, k2 = kb + 16 + i;       // k = c*16 + py*4 + px
      a[i]   = f2bf(x[(((size_t)(b << 8) + (k1 >> 4)) * 64 + hp * 4 + ((k1 >> 2) & 3)) * 128 + wp * 4 + (k1 & 3)]);
      a[i+8] = f2bf(x[(((size_t)(b << 8) + (k2 >> 4)) * 64 + hp * 4 + ((k2 >> 2) & 3)) * 128 + wp * 4 + (k2 & 3)]);
    }
    acc = wmma_bf(a, load_frag_b_pk(Bw, 128, nt, k0 >> 5, lane), acc);
  }
  float pb = patch_b[nt * 16 + (lane & 15)];
  bf16* D = t_bf + (size_t)b * 512 * 128 + (size_t)mt * 16 * 128 + nt * 16;
  int n = lane & 15, mb = (lane >> 4) << 3;
#pragma unroll
  for (int r = 0; r < 8; r++) D[(size_t)(mb + r) * 128 + n] = f2bf(acc[r] + pb);
}

// ---------------- stage 2: in_proj (per-batch M=512,N=512,K=128) -> xin / z ----------------
__global__ void k_in_proj(float* xin, float* zbuf, const bf16* t_bf, const bf16* Bw){
  int tile = blockIdx.x * 8 + WAVE;            // 8 * 32 * 32 = 8192 tiles
  int b = tile >> 10, mt = (tile >> 5) & 31, nt = tile & 31;
  int lane = LANE;
  const bf16* A = t_bf + (size_t)b * 512 * 128 + (size_t)mt * 16 * 128;
  v8f acc = {};
  for (int k0 = 0; k0 < 128; k0 += 32)
    acc = wmma_bf(load_frag_a(A, 128, k0, lane), load_frag_b_pk(Bw, 4, nt, k0 >> 5, lane), acc);
  int col = nt * 16 + (lane & 15);
  int mrow = mt * 16 + ((lane >> 4) << 3);
#pragma unroll
  for (int r = 0; r < 8; r++){
    size_t row = (size_t)b * 512 + mrow + r;
    if (col < 256) xin[row * 256 + col]          = acc[r];
    else           zbuf[row * 256 + (col - 256)] = acc[r];
  }
}

// ---------------- stage 3: causal depthwise conv1d + SiLU ----------------
__global__ void k_conv1d_silu(float* xc, bf16* xc_bf, const float* xin, const float* w, const float* bias){
  int i = blockIdx.x * 256 + threadIdx.x;
  if (i >= 8 * 512 * 256) return;
  int d = i & 255, n = (i >> 8) & 511, b = i >> 17;
  const float* base = xin + ((size_t)b * 512) * 256 + d;
  float s = bias[d];
#pragma unroll
  for (int k = 0; k < 4; k++){ int nn = n - 3 + k; if (nn >= 0) s += base[(size_t)nn * 256] * w[d * 4 + k]; }
  float v = silu_f(s);
  xc[i] = v; xc_bf[i] = f2bf(v);
}

// ---------------- stage 4: x_proj (per-batch M=512,N=48(pad 40),K=256) ----------------
__global__ void k_x_proj(float* dbl, const bf16* xc_bf, const bf16* Bw){
  int tile = blockIdx.x * 8 + WAVE;            // 8 * 32 * 3 = 768 tiles
  int b = tile / 96; int rem = tile - b * 96; int mt = rem / 3, nt = rem - mt * 3;
  int lane = LANE;
  const bf16* A = xc_bf + (size_t)b * 512 * 256 + (size_t)mt * 16 * 256;
  v8f acc = {};
  for (int k0 = 0; k0 < 256; k0 += 32)
    acc = wmma_bf(load_frag_a(A, 256, k0, lane), load_frag_b_pk(Bw, 8, nt, k0 >> 5, lane), acc);
  int col = nt * 16 + (lane & 15);
  int mrow = mt * 16 + ((lane >> 4) << 3);
  if (col < 40){
#pragma unroll
    for (int r = 0; r < 8; r++) dbl[((size_t)b * 512 + mrow + r) * 40 + col] = acc[r];
  }
}

// ---------------- stage 5: dt = softplus(dbl[:,:8] @ dt_w^T + b) ----------------
__global__ void k_dt(float* dtb, const float* dbl, const float* w, const float* bias){
  int i = blockIdx.x * 256 + threadIdx.x;
  if (i >= 8 * 512 * 256) return;
  int d = i & 255; size_t bn = (size_t)(i >> 8);
  const float* r = dbl + bn * 40;
  float s = bias[d];
#pragma unroll
  for (int j = 0; j < 8; j++) s += r[j] * w[d * 8 + j];
  dtb[i] = (s > 20.f) ? s : log1pf(__expf(s));
}

// ---------------- stage 6: selective scan, fused D*xc and silu(z) gate ----------------
__global__ void k_scan(bf16* y_bf, const float* dtb, const float* xc, const float* dbl,
                       const float* zbuf, const float* A_log, const float* D_param){
  int b = blockIdx.x >> 4;                                   // 16 blocks per batch
  int d = ((blockIdx.x & 15) * 8 + WAVE) * 2 + (LANE >> 4);  // 2 channels per wave
  int s = LANE & 15;                                         // 16 SSM states per channel
  float Av = -__expf(A_log[d * 16 + s]);
  float Dv = D_param[d];
  float h = 0.f;
  const float* dt_p = dtb  + (size_t)b * 512 * 256 + d;
  const float* xc_p = xc   + (size_t)b * 512 * 256 + d;
  const float* z_p  = zbuf + (size_t)b * 512 * 256 + d;
  const float* db_p = dbl  + (size_t)b * 512 * 40;
  bf16* y_p = y_bf + (size_t)b * 512 * 256 + d;
  for (int n = 0; n < 512; n++){
    float dt = dt_p[(size_t)n * 256];
    float xv = xc_p[(size_t)n * 256];
    float Bv = db_p[n * 40 + 8 + s];
    float Cv = db_p[n * 40 + 24 + s];
    h = __expf(dt * Av) * h + dt * Bv * xv;
    float p = h * Cv;
    p += __shfl_xor(p, 1); p += __shfl_xor(p, 2);
    p += __shfl_xor(p, 4); p += __shfl_xor(p, 8);
    if (s == 0){
      float zv = z_p[(size_t)n * 256];
      y_p[(size_t)n * 256] = f2bf((p + Dv * xv) * silu_f(zv));
    }
  }
}

// ---------------- stage 7: out_proj (per-batch M=512,N=128,K=256) ----------------
// D tile stored DIRECTLY in the lin-GEMM B-fragment layout (ktiles=16):
// kt = mt>>1, k-half = mt&1; each lane emits one contiguous 16B run.
__global__ void k_out_proj(bf16* pk_act, const bf16* y_bf, const bf16* Bw){
  int tile = blockIdx.x * 8 + WAVE;            // 8 * 32 * 8 = 2048 tiles
  int b = tile >> 8, mt = (tile >> 3) & 31, nt = tile & 7;
  int lane = LANE;
  const bf16* A = y_bf + (size_t)b * 512 * 256 + (size_t)mt * 16 * 256;
  v8f acc = {};
  for (int k0 = 0; k0 < 256; k0 += 32)
    acc = wmma_bf(load_frag_a(A, 256, k0, lane), load_frag_b_pk(Bw, 8, nt, k0 >> 5, lane), acc);
  bf16* p = pk_act + (size_t)b * 512 * 128
          + (((size_t)nt * 16 + (mt >> 1)) * 32 + lane) * 16 + (mt & 1) * 8;
#pragma unroll
  for (int r = 0; r < 8; r++) p[r] = f2bf(acc[r]);
}

// ---------------- stage 8: lin GEMM (M=8192,N=128,K=512 per batch), 64m x 16n per wave ----------------
__global__ void k_lin(bf16* u_bf, const bf16* lin_bf, const bf16* pk_act, const float* lin_b){
  int tile = blockIdx.x * 8 + WAVE;            // 8 * 128 * 8 = 8192 wave-tiles
  int b = tile >> 10, mg = (tile >> 3) & 127, nt = tile & 7;
  int lane = LANE;
  const bf16* A0 = lin_bf + (size_t)mg * 64 * 512;    // lin_w rows (shared over batch)
  const bf16* Bp = pk_act + (size_t)b * 512 * 128;    // frag-packed activations, ktiles=16
  v8f acc[4] = {{},{},{},{}};
  for (int k0 = 0; k0 < 512; k0 += 32){
    v16bf bfrag = load_frag_b_pk(Bp, 16, nt, k0 >> 5, lane);
    __builtin_prefetch(A0 + (size_t)(lane & 15) * 512 + k0 + 64, 0, 1);  // stream A panel
#pragma unroll
    for (int t = 0; t < 4; t++){
      v16bf a = load_frag_a(A0 + (size_t)t * 16 * 512, 512, k0, lane);
      acc[t] = wmma_bf(a, bfrag, acc[t]);
    }
  }
  int e = nt * 16 + (lane & 15);
#pragma unroll
  for (int t = 0; t < 4; t++){
    int mb = mg * 64 + t * 16 + ((lane >> 4) << 3);
#pragma unroll
    for (int r = 0; r < 8; r++){
      int m = mb + r;                          // m = hp*512 + wp*16 + py*4 + px
      float v = acc[t][r] + lin_b[m];
      int h = ((m >> 9) << 2) + ((m >> 2) & 3);
      int w = (((m >> 4) & 31) << 2) + (m & 3);
      u_bf[(((size_t)b * 128 + e) * 64 + h) * 128 + w] = f2bf(v);
    }
  }
}

// ---------------- stage 9: 3x3 conv (implicit GEMM, M=8192,N=512,K=1152 per batch) ----------------
// Block = one image row (128 spatial) x 64 oc (4 n-tiles). 247 KB LDS (CDNA5 WGP LDS = 320 KB):
//   sB: 4 fragment-packed weight panels (144 KB), loaded via TENSOR_LOAD_TO_LDS (TDM) when available
//   sA: channel-last halo tile [3 rows][130 cols][128 ch] (97.5 KB), zero-padded edges
// Wave: 16 spatial x 64 oc -> A fragment reused across 4 WMMAs.
__global__ void k_conv3x3(float* c_out, bf16* c_bf, const bf16* u_bf, const bf16* Bw, const float* conv_b){
  extern __shared__ bf16 smem[];
  bf16* sB = smem;                 // 4*36*512 = 73728 elems (147456 B)
  bf16* sA = smem + 73728;         // 49920 elems
  int b   = blockIdx.x >> 9;       // 8 * 64 * 8 = 4096 blocks
  int h   = (blockIdx.x >> 3) & 63;
  int nt0 = (blockIdx.x & 7) * 4;

#ifdef HAVE_TDM
  if (WAVE == 0){
    // Tensor DMA: 1-D descriptor, data_size=8B, 18432 units = 147456 bytes, global -> LDS.
    unsigned long long ga = (unsigned long long)(const void*)(Bw + (size_t)nt0 * 36 * 512);
    unsigned int lds_off = (unsigned int)(unsigned long long)(void*)sB; // LDS aperture low bits = byte offset
    v4u g0;
    g0[0] = 1u;                                                // count=1, user descriptor
    g0[1] = lds_off;                                           // lds_addr (bytes)
    g0[2] = (unsigned int)(ga & 0xFFFFFFFFu);                  // global_addr[31:0]
    g0[3] = (unsigned int)((ga >> 32) & 0x1FFFFFFu) | (2u << 30); // global_addr[56:32] | type=2
    v8i g1;
    g1[0] = (int)(3u << 16);                                   // data_size = 3 (8 bytes)
    g1[1] = (int)((18432u & 0xFFFFu) << 16);                   // tensor_dim0[15:0]
    g1[2] = (int)(18432u >> 16) | (int)(1u << 16);             // tensor_dim0[31:16] | tensor_dim1=1
    g1[3] = (int)((18432u & 0xFFFFu) << 16);                   // tile_dim0 = 18432
    g1[4] = 1;                                                 // tile_dim1 = 1
    g1[5] = 18432;                                             // tensor_dim0_stride
    g1[6] = 0; g1[7] = 0;
    v4i z = {};
#if __clang_major__ >= 23
    v8i z8 = {};
    __builtin_amdgcn_tensor_load_to_lds(g0, g1, z, z, z8, 0);
#else
    __builtin_amdgcn_tensor_load_to_lds(g0, g1, z, z, 0);
#endif
  }
#else
  { // fallback: contiguous 144 KB panel copy through VGPRs
    const uint4* src4 = (const uint4*)(Bw + (size_t)nt0 * 36 * 512);
    uint4* dst4 = (uint4*)sB;
    for (int j = threadIdx.x; j < 9216; j += 256) dst4[j] = src4[j];
  }
#endif

  { // stage sA: u[b, c, h-1..h+1, -1..128] channel-last with zero pad (overlaps with TDM)
    const bf16 zero = f2bf(0.f);
    for (int idx = threadIdx.x; idx < 128 * 390; idx += 256){
      int c = idx / 390; int r = idx - c * 390;
      int ky = r / 130;  int j = r - ky * 130;
      int hy = h + ky - 1, wx = j - 1;
      bf16 v = (hy >= 0 && hy < 64 && wx >= 0 && wx < 128)
             ? u_bf[(((size_t)b * 128 + c) * 64 + hy) * 128 + wx] : zero;
      sA[((size_t)ky * 130 + j) * 128 + c] = v;
    }
  }
#ifdef HAVE_TDM
  if (WAVE == 0) __builtin_amdgcn_s_wait_tensorcnt(0);
#endif
  __syncthreads();

  int lane = LANE;
  int w = WAVE * 16 + (lane & 15); // this wave's 16 spatial columns within the row
  v8f acc[4] = {{},{},{},{}};
  for (int k0 = 0; k0 < 1152; k0 += 32){
    int pos = k0 >> 7;             // uniform filter tap for this K-chunk
    int ky = pos / 3, kx = pos - ky * 3;
    int cb = (k0 & 127) + ((lane >> 4) << 3);
    const bf16* pa = sA + ((size_t)ky * 130 + w + kx) * 128 + cb;  // contiguous in channel
    v16bf a;
#pragma unroll
    for (int i = 0; i < 8; i++){ a[i] = pa[i]; a[i+8] = pa[16 + i]; }
    int kt = k0 >> 5;
#pragma unroll
    for (int t = 0; t < 4; t++){
      const bf16* pb = sB + (((size_t)t * 36 + kt) * 32 + lane) * 16;
      v16bf bb;
#pragma unroll
      for (int i = 0; i < 16; i++) bb[i] = pb[i];
      acc[t] = wmma_bf(a, bb, acc[t]);
    }
  }

  int wb = WAVE * 16 + ((lane >> 4) << 3);
#pragma unroll
  for (int t = 0; t < 4; t++){
    int oc = (nt0 + t) * 16 + (lane & 15);
    int g = oc >> 2, sy = (oc >> 1) & 1, sx = oc & 1;
    float cbv = conv_b[oc];
#pragma unroll
    for (int r = 0; r < 8; r++){
      int wr = wb + r;
      int h2 = 2 * h + sy, w2 = 2 * wr + sx;
      float v = gelu_f(acc[t][r] + cbv);
      size_t hw2 = (size_t)h2 * 256 + w2;
      c_out[((size_t)b * 128 + g) * 32768 + hw2] = v;          // NCHW fp32 (residual base)
      c_bf[((size_t)b * 32768 + hw2) * 128 + g]  = f2bf(v);    // channel-last bf16 for FFN
    }
  }
}

// ---------------- stage 10/11: per-pixel FFN (M=262144, N=128, K=128), 64m x 16n per wave ----------------
__global__ void k_ffn1(bf16* f_bf, const bf16* c_bf, const bf16* Bw, const float* b1){
  int tile = blockIdx.x * 8 + WAVE;            // 4096 mg * 8 nt = 32768 wave-tiles
  int mg = tile >> 3, nt = tile & 7;
  int lane = LANE;
  const bf16* A0 = c_bf + (size_t)mg * 64 * 128;
  v8f acc[4] = {{},{},{},{}};
  for (int k0 = 0; k0 < 128; k0 += 32){
    v16bf bfrag = load_frag_b_pk(Bw, 4, nt, k0 >> 5, lane);
#pragma unroll
    for (int t = 0; t < 4; t++)
      acc[t] = wmma_bf(load_frag_a(A0 + (size_t)t * 16 * 128, 128, k0, lane), bfrag, acc[t]);
  }
  int e = nt * 16 + (lane & 15);
  float bias = b1[e];
  int mbase = mg * 64 + ((lane >> 4) << 3);
#pragma unroll
  for (int t = 0; t < 4; t++)
#pragma unroll
    for (int r = 0; r < 8; r++)
      f_bf[(size_t)(mbase + t * 16 + r) * 128 + e] = f2bf(gelu_f(acc[t][r] + bias));
}

__global__ void k_ffn2_residual(float* c_out, const bf16* f_bf, const bf16* Bw, const float* b2){
  int tile = blockIdx.x * 8 + WAVE;
  int mg = tile >> 3, nt = tile & 7;
  int lane = LANE;
  const bf16* A0 = f_bf + (size_t)mg * 64 * 128;
  v8f acc[4] = {{},{},{},{}};
  for (int k0 = 0; k0 < 128; k0 += 32){
    v16bf bfrag = load_frag_b_pk(Bw, 4, nt, k0 >> 5, lane);
#pragma unroll
    for (int t = 0; t < 4; t++)
      acc[t] = wmma_bf(load_frag_a(A0 + (size_t)t * 16 * 128, 128, k0, lane), bfrag, acc[t]);
  }
  int c = nt * 16 + (lane & 15);
  float bias = b2[c];
  int mbase = mg * 64 + ((lane >> 4) << 3);
#pragma unroll
  for (int t = 0; t < 4; t++)
#pragma unroll
    for (int r = 0; r < 8; r++){
      int m = mbase + t * 16 + r;              // m = b*32768 + hw2
      int b = m >> 15; size_t hw2 = (size_t)(m & 32767);
      float* o = c_out + ((size_t)b * 128 + c) * 32768 + hw2;
      *o = *o + acc[t][r] + bias;              // out = c + f
    }
}

// ---------------- host launcher ----------------
extern "C" void kernel_launch(void* const* d_in, const int* /*in_sizes*/, int /*n_in*/,
                              void* d_out, int /*out_size*/, void* d_ws, size_t /*ws_size*/,
                              hipStream_t stream){
  const float* x         = (const float*)d_in[0];
  const float* patch_w   = (const float*)d_in[1];
  const float* patch_b   = (const float*)d_in[2];
  const float* in_proj_w = (const float*)d_in[3];
  const float* conv1d_w  = (const float*)d_in[4];
  const float* conv1d_b  = (const float*)d_in[5];
  const float* x_proj_w  = (const float*)d_in[6];
  const float* dt_proj_w = (const float*)d_in[7];
  const float* dt_proj_b = (const float*)d_in[8];
  const float* A_log     = (const float*)d_in[9];
  const float* D_param   = (const float*)d_in[10];
  const float* out_proj_w= (const float*)d_in[11];
  const float* lin_w     = (const float*)d_in[12];
  const float* lin_b     = (const float*)d_in[13];
  const float* conv_w    = (const float*)d_in[14];
  const float* conv_b    = (const float*)d_in[15];
  const float* ffn_w1    = (const float*)d_in[16];
  const float* ffn_b1    = (const float*)d_in[17];
  const float* ffn_w2    = (const float*)d_in[18];
  const float* ffn_b2    = (const float*)d_in[19];
  float* out = (float*)d_out;

  size_t off = 0;
  auto carve = [&](size_t bytes) -> void* {
    off = (off + 255) & ~(size_t)255;
    void* p = (char*)d_ws + off; off += bytes; return p;
  };
  bf16* pk_patch   = (bf16*)carve((size_t)4096 * 128 * 2);
  bf16* pk_inproj  = (bf16*)carve((size_t)128 * 512 * 2);
  bf16* pk_xproj   = (bf16*)carve((size_t)256 * 48 * 2);
  bf16* pk_outproj = (bf16*)carve((size_t)256 * 128 * 2);
  bf16* pk_conv    = (bf16*)carve((size_t)1152 * 512 * 2);
  bf16* pk_f1      = (bf16*)carve((size_t)128 * 128 * 2);
  bf16* pk_f2      = (bf16*)carve((size_t)128 * 128 * 2);
  bf16* lin_bf     = (bf16*)carve((size_t)8192 * 512 * 2);
  bf16* t_bf       = (bf16*)carve((size_t)8 * 512 * 128 * 2);
  float* xin       = (float*)carve((size_t)8 * 512 * 256 * 4);
  float* zbuf      = (float*)carve((size_t)8 * 512 * 256 * 4);
  float* xc        = (float*)carve((size_t)8 * 512 * 256 * 4);
  bf16*  xc_bf     = (bf16*) carve((size_t)8 * 512 * 256 * 2);
  float* dbl       = (float*)carve((size_t)8 * 512 * 40 * 4);
  float* dtb       = (float*)carve((size_t)8 * 512 * 256 * 4);
  bf16*  y_bf      = (bf16*) carve((size_t)8 * 512 * 256 * 2);
  bf16*  pk_act    = (bf16*) carve((size_t)8 * 512 * 128 * 2);   // frag-packed out_proj result
  bf16*  u_bf      = (bf16*) carve((size_t)8 * 128 * 64 * 128 * 2);
  bf16*  c_bf      = (bf16*) carve((size_t)8 * 32768 * 128 * 2);
  bf16*  f_bf      = (bf16*) carve((size_t)8 * 32768 * 128 * 2);

  auto blocks = [](size_t n){ return (int)((n + 255) / 256); };

  // weight packing into fragment layouts (bf16)
  k_pack_fragB<<<blocks((size_t)4096*128), 256, 0, stream>>>(pk_patch,   patch_w,   4096, 4096, 128, 128);
  k_pack_fragB<<<blocks((size_t)128*512),  256, 0, stream>>>(pk_inproj,  in_proj_w, 128,  128,  512, 512);
  k_pack_fragB<<<blocks((size_t)256*48),   256, 0, stream>>>(pk_xproj,   x_proj_w,  256,  256,  48,  40);
  k_pack_fragB<<<blocks((size_t)256*128),  256, 0, stream>>>(pk_outproj, out_proj_w,256,  256,  128, 128);
  k_pack_fragB<<<blocks((size_t)128*128),  256, 0, stream>>>(pk_f1,      ffn_w1,    128,  128,  128, 128);
  k_pack_fragB<<<blocks((size_t)128*128),  256, 0, stream>>>(pk_f2,      ffn_w2,    128,  128,  128, 128);
  k_pack_conv_fragB<<<blocks((size_t)1152*512), 256, 0, stream>>>(pk_conv, conv_w);
  k_convert_bf16<<<blocks((size_t)8192*512), 256, 0, stream>>>(lin_bf, lin_w, 8192*512);

  // pipeline
  k_patch_embed <<<256,  256, 0, stream>>>(t_bf, x, pk_patch, patch_b);
  k_in_proj     <<<1024, 256, 0, stream>>>(xin, zbuf, t_bf, pk_inproj);
  k_conv1d_silu <<<blocks((size_t)8*512*256), 256, 0, stream>>>(xc, xc_bf, xin, conv1d_w, conv1d_b);
  k_x_proj      <<<96,   256, 0, stream>>>(dbl, xc_bf, pk_xproj);
  k_dt          <<<blocks((size_t)8*512*256), 256, 0, stream>>>(dtb, dbl, dt_proj_w, dt_proj_b);
  k_scan        <<<128,  256, 0, stream>>>(y_bf, dtb, xc, dbl, zbuf, A_log, D_param);
  k_out_proj    <<<256,  256, 0, stream>>>(pk_act, y_bf, pk_outproj);
  k_lin         <<<1024, 256, 0, stream>>>(u_bf, lin_bf, pk_act, lin_b);

  size_t conv_lds = (size_t)(73728 + 49920) * sizeof(bf16);   // 247 KB (CDNA5 WGP LDS = 320 KB)
  k_conv3x3     <<<4096, 256, conv_lds, stream>>>(out, c_bf, u_bf, pk_conv, conv_b);

  k_ffn1        <<<4096, 256, 0, stream>>>(f_bf, c_bf, pk_f1, ffn_b1);
  k_ffn2_residual<<<4096, 256, 0, stream>>>(out, f_bf, pk_f2, ffn_b2);
}
